// MultiheadAttention_19481971655349
// MI455X (gfx1250) — matrix operand order repaired and checked by reading, not compiled, not run on hardware
//
#include <hip/hip_runtime.h>
#include <hip/hip_bf16.h>

// ---------------------------------------------------------------------------
// Multi-head causal attention, bf16 WMMA path for gfx1250 (MI455X).
//   S=4096 seq, D=1024 model, H=16 heads, hd=64.
//   - All GEMMs via v_wmma_f32_16x16x32_bf16, 16x64 register blocking.
//   - Flash-style online softmax, 64-key blocks; row-sums via WMMA x ones.
//   - Causal mask analytic, applied only on diagonal blocks.
//   - __launch_bounds__(256,1) so the VGPR budget fits without spills.
// ---------------------------------------------------------------------------

typedef __bf16 bf16_t;
typedef __attribute__((ext_vector_type(16))) __bf16     v16bf;
typedef __attribute__((ext_vector_type(8)))  float       v8f;
typedef __attribute__((ext_vector_type(4)))  unsigned int v4u;

constexpr int SEQ    = 4096;
constexpr int DMODEL = 1024;
constexpr int NH     = 16;
constexpr int HDIM   = 64;
constexpr int PPITCH = 72;   // LDS pitch (bf16) for 16x64 P tile; 144B = 9*16B

union FragCast { v4u u[2]; v16bf v; };

// A-fragment (16x32 bf16, row-major source, K contiguous per row).
// lane l: row = l&15 ; K = kb..kb+7 and kb+16..kb+23, kb = (l<16)?0:8.
static __device__ inline v16bf load_A_frag(const bf16_t* tile, int ld, int lane) {
  const int r  = lane & 15;
  const int kb = (lane < 16) ? 0 : 8;
  const bf16_t* p = tile + r * ld + kb;
  FragCast f;
  f.u[0] = *reinterpret_cast<const v4u*>(p);
  f.u[1] = *reinterpret_cast<const v4u*>(p + 16);
  return f.v;
}

// B-fragment (32x16 bf16) from an n-major matrix (row n, K contiguous).
// lane l: col = l&15 ; K = kb..kb+7 and kb+16..kb+23, kb = (l<16)?0:8.
static __device__ inline v16bf load_B_frag(const bf16_t* tile, int ld, int lane) {
  const int n  = lane & 15;
  const int kb = (lane < 16) ? 0 : 8;
  const bf16_t* p = tile + n * ld + kb;
  FragCast f;
  f.u[0] = *reinterpret_cast<const v4u*>(p);
  f.u[1] = *reinterpret_cast<const v4u*>(p + 16);
  return f.v;
}

static __device__ inline v16bf ones_B_frag() {
  FragCast f;
  const v4u o = {0x3F803F80u, 0x3F803F80u, 0x3F803F80u, 0x3F803F80u};  // bf16 1.0
  f.u[0] = o; f.u[1] = o;
  return f.v;
}

static __device__ inline v8f wmma_bf16(v16bf a, v16bf b, v8f c) {
  return __builtin_amdgcn_wmma_f32_16x16x32_bf16(false, a, false, b, (short)0, c,
                                                 false, false);
}

// ---------------------------------------------------------------------------
// fp32 -> bf16 cast, 4 elements / thread.
// ---------------------------------------------------------------------------
__global__ void __launch_bounds__(256)
cvt_f32_bf16(const float* __restrict__ in, bf16_t* __restrict__ out, int n) {
  const int i = (blockIdx.x * blockDim.x + threadIdx.x) * 4;
  if (i + 3 < n) {
    const float4 v = *reinterpret_cast<const float4*>(in + i);
    union { bf16_t b[4]; unsigned long long q; } u;
    u.b[0] = (bf16_t)v.x; u.b[1] = (bf16_t)v.y;
    u.b[2] = (bf16_t)v.z; u.b[3] = (bf16_t)v.w;
    *reinterpret_cast<unsigned long long*>(out + i) = u.q;
  }
}

// ---------------------------------------------------------------------------
// y = A(bf16)[SEQ,DMODEL] @ W(bf16)[DMODEL,DMODEL]^T + bias.
// Each wave computes a 16x64 strip (4 accumulators share one A-fragment).
// MODE 0: bf16 head-major [H][SEQ][64]            (K)
// MODE 1: bf16 transposed [H][64][SEQ]            (V)
// MODE 2: fp32 [SEQ][DMODEL]                      (output projection)
// MODE 3: bf16 head-major [H][SEQ][64], *0.125    (Q, softmax scale folded)
// ---------------------------------------------------------------------------
template <int MODE>
__global__ void __launch_bounds__(256, 1)
gemm16(const bf16_t* __restrict__ A, const bf16_t* __restrict__ W,
       const float* __restrict__ bias, bf16_t* __restrict__ outb,
       float* __restrict__ outf) {
  const int lane  = threadIdx.x & 31;
  const int wave  = threadIdx.x >> 5;
  const int mtile = blockIdx.x;
  const int nt0   = (blockIdx.y * 8 + wave) * 4;   // first of 4 N-tiles

  const bf16_t* arow = A + (size_t)mtile * 16 * DMODEL;
  const bf16_t* brow = W + (size_t)nt0 * 16 * DMODEL;

  v8f acc[4] = {{}, {}, {}, {}};
#pragma unroll 2
  for (int k0 = 0; k0 < DMODEL; k0 += 32) {
    const v16bf a = load_A_frag(arow + k0, DMODEL, lane);
#pragma unroll
    for (int t = 0; t < 4; ++t) {
      const v16bf b = load_B_frag(brow + (size_t)t * 16 * DMODEL + k0, DMODEL, lane);
      acc[t] = wmma_bf16(a, b, acc[t]);
    }
  }

  const int moff = (lane < 16) ? 0 : 8;
  const int ncol = lane & 15;
#pragma unroll
  for (int t = 0; t < 4; ++t) {
    const int   n  = (nt0 + t) * 16 + ncol;
    const float bv = bias[n];
    if (MODE == 0 || MODE == 3) {
      const int h = n >> 6, d = n & 63;
      bf16_t* dst = outb + ((size_t)h * SEQ) * HDIM + d;
#pragma unroll
      for (int j = 0; j < 8; ++j) {
        const int s = mtile * 16 + moff + j;
        float v = acc[t][j] + bv;
        if (MODE == 3) v *= 0.125f;
        dst[(size_t)s * HDIM] = (bf16_t)v;
      }
    } else if (MODE == 1) {
      const int h = n >> 6, d = n & 63;
      bf16_t* dst = outb + ((size_t)h * HDIM + d) * SEQ;
#pragma unroll
      for (int j = 0; j < 8; ++j) {
        const int s = mtile * 16 + moff + j;
        dst[s] = (bf16_t)(acc[t][j] + bv);
      }
    } else {
#pragma unroll
      for (int j = 0; j < 8; ++j) {
        const int s = mtile * 16 + moff + j;
        outf[(size_t)s * DMODEL + n] = acc[t][j] + bv;
      }
    }
  }
}

// ---------------------------------------------------------------------------
// Flash attention: one head per blockIdx.y, 128 queries per block
// (8 waves x 16 queries), 64 keys per inner iteration.
// Q (pre-scaled by 1/8), K: [H][SEQ][64] bf16.  VT: [H][64][SEQ] bf16.
// Z: [SEQ][DMODEL] bf16.
// ---------------------------------------------------------------------------
__global__ void __launch_bounds__(256, 1)
attn_fwd(const bf16_t* __restrict__ Q, const bf16_t* __restrict__ K,
         const bf16_t* __restrict__ VT, bf16_t* __restrict__ Z) {
  __shared__ __align__(16) bf16_t pbuf[8 * 16 * PPITCH];  // per-wave 16x64 P tile

  const int lane  = threadIdx.x & 31;
  const int wave  = threadIdx.x >> 5;
  const int h     = blockIdx.y;
  const int qbase = blockIdx.x * 128 + wave * 16;
  bf16_t* pw = pbuf + wave * 16 * PPITCH;

  const bf16_t* qtile = Q + ((size_t)h * SEQ + qbase) * HDIM;
  const v16bf aq0 = load_A_frag(qtile, HDIM, lane);
  const v16bf aq1 = load_A_frag(qtile + 32, HDIM, lane);
  const v16bf bones = ones_B_frag();

  float mrow[8], lrow[8];
  v8f o[4] = {{}, {}, {}, {}};
#pragma unroll
  for (int j = 0; j < 8; ++j) { mrow[j] = -3.0e38f; lrow[j] = 0.0f; }

  const int moff = (lane < 16) ? 0 : 8;
  const int ncol = lane & 15;
  // per-row mask threshold: column c of tile t is masked iff kb + 16t > qmr[j]
  int qmr[8];
#pragma unroll
  for (int j = 0; j < 8; ++j) qmr[j] = qbase + moff + j - ncol;

  const int kend = qbase + 16;  // need keys k <= qbase+15

  for (int kb = 0; kb < kend; kb += 64) {
    // ---- scores: Q[16x64] x K^T -> four 16x16 f32 tiles (8 WMMA) ----
    const bf16_t* kt = K + ((size_t)h * SEQ + kb) * HDIM;
    v8f s[4];
#pragma unroll
    for (int t = 0; t < 4; ++t) {
      v8f st = {};
      st = wmma_bf16(aq0, load_B_frag(kt + t * 16 * HDIM,      HDIM, lane), st);
      st = wmma_bf16(aq1, load_B_frag(kt + t * 16 * HDIM + 32, HDIM, lane), st);
      s[t] = st;
    }

    // ---- causal mask (diagonal blocks only) + per-row local max ----
    float xm[8];
    if (kb + 63 > qbase) {
#pragma unroll
      for (int j = 0; j < 8; ++j) {
        float x0 = (kb      > qmr[j]) ? -1.0e30f : s[0][j];
        float x1 = (kb + 16 > qmr[j]) ? -1.0e30f : s[1][j];
        float x2 = (kb + 32 > qmr[j]) ? -1.0e30f : s[2][j];
        float x3 = (kb + 48 > qmr[j]) ? -1.0e30f : s[3][j];
        s[0][j] = x0; s[1][j] = x1; s[2][j] = x2; s[3][j] = x3;
        xm[j] = fmaxf(fmaxf(x0, x1), fmaxf(x2, x3));
      }
    } else {
#pragma unroll
      for (int j = 0; j < 8; ++j)
        xm[j] = fmaxf(fmaxf(s[0][j], s[1][j]), fmaxf(s[2][j], s[3][j]));
    }
    // batched butterfly max across the 16-lane halves (8 bpermutes per level)
#pragma unroll
    for (int lvl = 1; lvl < 16; lvl <<= 1) {
      float tt[8];
#pragma unroll
      for (int j = 0; j < 8; ++j) tt[j] = __shfl_xor(xm[j], lvl);
#pragma unroll
      for (int j = 0; j < 8; ++j) xm[j] = fmaxf(xm[j], tt[j]);
    }

    // ---- exp, accumulator rescale, P -> LDS (bf16) ----
    float corr[8];
#pragma unroll
    for (int j = 0; j < 8; ++j) {
      const float mnew = fmaxf(mrow[j], xm[j]);
      const float c    = __expf(mrow[j] - mnew);
      mrow[j] = mnew; corr[j] = c;
      o[0][j] *= c; o[1][j] *= c; o[2][j] *= c; o[3][j] *= c;
      const float p0 = __expf(s[0][j] - mnew);
      const float p1 = __expf(s[1][j] - mnew);
      const float p2 = __expf(s[2][j] - mnew);
      const float p3 = __expf(s[3][j] - mnew);
      bf16_t* pr = pw + (moff + j) * PPITCH + ncol;
      pr[0]  = (bf16_t)p0;
      pr[16] = (bf16_t)p1;
      pr[32] = (bf16_t)p2;
      pr[48] = (bf16_t)p3;
    }

    // ---- P fragments; row-sum via WMMA against ones (replaces shuffles) ----
    const v16bf ap0 = load_A_frag(pw,      PPITCH, lane);
    const v16bf ap1 = load_A_frag(pw + 32, PPITCH, lane);
    v8f bs = {};
    bs = wmma_bf16(ap0, bones, bs);
    bs = wmma_bf16(ap1, bones, bs);
#pragma unroll
    for (int j = 0; j < 8; ++j) lrow[j] = lrow[j] * corr[j] + bs[j];

    // ---- PV: P[16x64] x V[64x64] accumulated into o[0..3] (8 WMMA) ----
    const bf16_t* vt = VT + ((size_t)h * HDIM) * SEQ + kb;
#pragma unroll
    for (int t = 0; t < 4; ++t) {
      const v16bf bv0 = load_B_frag(vt + (size_t)t * 16 * SEQ,      SEQ, lane);
      const v16bf bv1 = load_B_frag(vt + (size_t)t * 16 * SEQ + 32, SEQ, lane);
      o[t] = wmma_bf16(ap0, bv0, o[t]);
      o[t] = wmma_bf16(ap1, bv1, o[t]);
    }
  }

  // ---- normalize and write Z [SEQ][DMODEL] bf16 ----
#pragma unroll
  for (int j = 0; j < 8; ++j) {
    const float inv = 1.0f / lrow[j];
    const int s = qbase + moff + j;
    bf16_t* zp = Z + (size_t)s * DMODEL + h * HDIM + ncol;
    zp[0]  = (bf16_t)(o[0][j] * inv);
    zp[16] = (bf16_t)(o[1][j] * inv);
    zp[32] = (bf16_t)(o[2][j] * inv);
    zp[48] = (bf16_t)(o[3][j] * inv);
  }
}

// ---------------------------------------------------------------------------
extern "C" void kernel_launch(void* const* d_in, const int* in_sizes, int n_in,
                              void* d_out, int out_size, void* d_ws, size_t ws_size,
                              hipStream_t stream) {
  (void)in_sizes; (void)n_in; (void)out_size; (void)ws_size;

  const float* x    = (const float*)d_in[0];
  // d_in[1] is the causal bool mask: computed analytically instead.
  const float* wq_w = (const float*)d_in[2];
  const float* wq_b = (const float*)d_in[3];
  const float* wk_w = (const float*)d_in[4];
  const float* wk_b = (const float*)d_in[5];
  const float* wv_w = (const float*)d_in[6];
  const float* wv_b = (const float*)d_in[7];
  const float* wo_w = (const float*)d_in[8];
  const float* wo_b = (const float*)d_in[9];
  float* out = (float*)d_out;

  const size_t NX = (size_t)SEQ * DMODEL;      // 4M
  const size_t NW = (size_t)DMODEL * DMODEL;   // 1M
  bf16_t* ws  = (bf16_t*)d_ws;
  bf16_t* xb  = ws;            // [SEQ][D]
  bf16_t* wqb = xb  + NX;      // [D][D]
  bf16_t* wkb = wqb + NW;
  bf16_t* wvb = wkb + NW;
  bf16_t* wob = wvb + NW;
  bf16_t* Qh  = wob + NW;      // [H][SEQ][64] (pre-scaled by 1/8)
  bf16_t* Kh  = Qh  + NX;      // [H][SEQ][64]
  bf16_t* Vt  = Kh  + NX;      // [H][64][SEQ]
  bf16_t* Zb  = Vt  + NX;      // [SEQ][D]      total 24M bf16 = 48 MB

  const dim3 blk(256);
  cvt_f32_bf16<<<dim3((unsigned)(NX / 1024)), blk, 0, stream>>>(x,    xb,  (int)NX);
  cvt_f32_bf16<<<dim3((unsigned)(NW / 1024)), blk, 0, stream>>>(wq_w, wqb, (int)NW);
  cvt_f32_bf16<<<dim3((unsigned)(NW / 1024)), blk, 0, stream>>>(wk_w, wkb, (int)NW);
  cvt_f32_bf16<<<dim3((unsigned)(NW / 1024)), blk, 0, stream>>>(wv_w, wvb, (int)NW);
  cvt_f32_bf16<<<dim3((unsigned)(NW / 1024)), blk, 0, stream>>>(wo_w, wob, (int)NW);

  // 16x64 strip per wave: grid (SEQ/16, DMODEL/(8 waves * 64 cols))
  const dim3 ggrid(SEQ / 16, DMODEL / 512);
  gemm16<3><<<ggrid, blk, 0, stream>>>(xb, wqb, wq_b, Qh, nullptr);
  gemm16<0><<<ggrid, blk, 0, stream>>>(xb, wkb, wk_b, Kh, nullptr);
  gemm16<1><<<ggrid, blk, 0, stream>>>(xb, wvb, wv_b, Vt, nullptr);

  attn_fwd<<<dim3(SEQ / 128, NH), blk, 0, stream>>>(Qh, Kh, Vt, Zb);

  gemm16<2><<<ggrid, blk, 0, stream>>>(Zb, wob, wo_b, nullptr, out);
}
